// TemporalGridAttention_26577257628224
// MI455X (gfx1250) — compile-verified, matrix operand order checked by reading, not run on hardware
//
#include <hip/hip_runtime.h>
#include <hip/hip_bf16.h>
#include <math.h>

// ---- problem constants -----------------------------------------------------
#define GRID_N   30
#define S_N      900            // GRID*GRID
#define D_N      512
#define HQ_N     8
#define HKV_N    2
#define DH_N     64
#define T_N      16
#define NTOK     15301          // 1 + 16*900 + 900
#define NTP      15328          // NTOK padded to multiple of 32 (32*479)
#define QROWS    912            // 900 padded to multiple of 16 (16*57)
#define HIST_END 14400          // tokens 1..14400 are history
#define CUR_BASE 14401          // tokens 14401..15300 are current frame

typedef _Float16 v16h __attribute__((ext_vector_type(16)));
typedef _Float16 v8h  __attribute__((ext_vector_type(8)));
typedef float    v8f  __attribute__((ext_vector_type(8)));

// ---- WMMA tile helpers -----------------------------------------------------
// 16x32 f16 A/B operand from a row-major (row, k-contiguous) matrix.
// Lane l: r = l%16, half = l/16. VGPRs 0-3 hold K = half*8 + 0..7,
// VGPRs 4-7 hold K = 16 + half*8 + 0..7  (per CDNA5 ISA 7.12.2).
__device__ __forceinline__ v16h load_tile_AB(const _Float16* base, int ld) {
  int l = threadIdx.x & 31;
  int r = l & 15, half = l >> 4;
  const _Float16* p = base + (long)r * ld + half * 8;
  v16h out;
  *(v8h*)&out        = *(const v8h*)(p);
  *((v8h*)&out + 1)  = *(const v8h*)(p + 16);
  return out;
}

__device__ __forceinline__ v8f wmma_f16(v16h a, v16h b, v8f c) {
  return __builtin_amdgcn_wmma_f32_16x16x32_f16(false, a, false, b, (short)0, c, false, false);
}

// C/D 16x16 f32 layout: lanes 0-15 hold rows 0-7 (vgpr j = row j), lanes 16-31
// hold rows 8-15; lane%16 = column.
__device__ __forceinline__ void store_tile_C(float* base, int ld, v8f c, int rows_valid) {
  int l = threadIdx.x & 31;
  int col = l & 15;
  int rowbase = (l >> 4) * 8;
#pragma unroll
  for (int j = 0; j < 8; ++j) {
    int row = rowbase + j;
    if (row < rows_valid) base[(long)row * ld + col] = c[j];
  }
}

// cross-lane reductions within the 16-lane half-group (wave32)
__device__ __forceinline__ float redmax16(float v) {
  v = fmaxf(v, __shfl_xor(v, 1, 32));
  v = fmaxf(v, __shfl_xor(v, 2, 32));
  v = fmaxf(v, __shfl_xor(v, 4, 32));
  v = fmaxf(v, __shfl_xor(v, 8, 32));
  return v;
}
__device__ __forceinline__ float redsum16(float v) {
  v += __shfl_xor(v, 1, 32);
  v += __shfl_xor(v, 2, 32);
  v += __shfl_xor(v, 4, 32);
  v += __shfl_xor(v, 8, 32);
  return v;
}

// ---- packing kernels -------------------------------------------------------
__global__ void pack_tokens(const float* __restrict__ sg, const float* __restrict__ hist,
                            const float* __restrict__ init, _Float16* __restrict__ tok16) {
  long i = (long)blockIdx.x * blockDim.x + threadIdx.x;
  if (i >= (long)NTP * D_N) return;
  int t = (int)(i >> 9);          // /512
  int d = (int)(i & 511);
  float val;
  if (t == 0)                val = init[d];
  else if (t <= HIST_END)    val = hist[(long)(t - 1) * D_N + d];
  else if (t <  CUR_BASE + S_N) val = sg[(long)(t - CUR_BASE) * D_N + d];
  else                       val = 0.f;
  tok16[i] = (_Float16)val;
}

__global__ void cvt_f32_f16(const float* __restrict__ in, _Float16* __restrict__ o, int n) {
  int i = blockIdx.x * blockDim.x + threadIdx.x;
  if (i < n) o[i] = (_Float16)in[i];
}

// ---- generic WMMA GEMM: C(MxN,f32) = A16(Mpad x K) * B16(N x K)^T ---------
__global__ void __launch_bounds__(128)
gemm_f16(const _Float16* __restrict__ A, int lda,
         const _Float16* __restrict__ B, int ldb,
         float* __restrict__ C, int ldc,
         int Mtiles, int Ntiles, int K, int Mvalid) {
  int wave = blockIdx.x * (blockDim.x >> 5) + (threadIdx.x >> 5);
  if (wave >= Mtiles * Ntiles) return;
  int mt = wave / Ntiles, nt = wave % Ntiles;
  const _Float16* Ab = A + (long)mt * 16 * lda;
  const _Float16* Bb = B + (long)nt * 16 * ldb;
  v8f acc = {};
  for (int k0 = 0; k0 < K; k0 += 32) {
    v16h a = load_tile_AB(Ab + k0, lda);
    v16h b = load_tile_AB(Bb + k0, ldb);
    acc = wmma_f16(a, b, acc);
  }
  store_tile_C(C + (long)(mt * 16) * ldc + nt * 16, ldc, acc, Mvalid - mt * 16);
}

// ---- RoPE + f16 pack for K and transposed V --------------------------------
__global__ void rope_pack_kv(const float* __restrict__ kf, const float* __restrict__ vf,
                             _Float16* __restrict__ k16, _Float16* __restrict__ vT) {
  int i = blockIdx.x * blockDim.x + threadIdx.x;
  if (i >= NTP * 64) return;                 // 64 pairs per token (2 kv heads x 32)
  int t   = i >> 6;
  int pp  = i & 63;
  int kvh = pp >> 5;
  int pr  = pp & 31;                         // pair index within head, 0..31
  int base = t * 128 + kvh * DH_N + pr * 2;
  int dA   = kvh * DH_N + pr * 2;            // row in vT
  if (t >= NTOK) {                           // padded tokens -> zeros
    k16[base] = (_Float16)0.f; k16[base + 1] = (_Float16)0.f;
    vT[(long)dA * NTP + t] = (_Float16)0.f;
    vT[(long)(dA + 1) * NTP + t] = (_Float16)0.f;
    return;
  }
  float pos, cy, cx;
  if (t == 0) { pos = 0.f; cy = 0.f; cx = 0.f; }
  else if (t <= HIST_END) {
    int tt = (t - 1) / S_N, cell = (t - 1) % S_N;
    pos = (float)tt; cy = (float)(cell / GRID_N); cx = (float)(cell % GRID_N);
  } else {
    int cell = t - CUR_BASE;
    pos = (float)T_N; cy = (float)(cell / GRID_N); cx = (float)(cell % GRID_N);
  }
  // successive pairwise rotations compose: 1D(pos) then 2D(y|x)
  float ang = pos * __powf(10000.f, -(float)pr * (1.f / 32.f));
  if (pr < 16) ang += cy * __powf(10000.f, -(float)pr * (1.f / 16.f));
  else         ang += cx * __powf(10000.f, -(float)(pr - 16) * (1.f / 16.f));
  float s, c; __sincosf(ang, &s, &c);
  float k0 = kf[base], k1 = kf[base + 1];
  k16[base]     = (_Float16)(k0 * c - k1 * s);
  k16[base + 1] = (_Float16)(k0 * s + k1 * c);
  vT[(long)dA * NTP + t]       = (_Float16)vf[base];
  vT[(long)(dA + 1) * NTP + t] = (_Float16)vf[base + 1];
}

// ---- RoPE + 1/sqrt(DH) scale + f16 pack for Q ------------------------------
__global__ void rope_pack_q(const float* __restrict__ qf, _Float16* __restrict__ q16) {
  int i = blockIdx.x * blockDim.x + threadIdx.x;
  if (i >= QROWS * 256) return;              // 8 heads x 32 pairs
  int qr = i >> 8;
  int hp = i & 255;
  int h  = hp >> 5, pr = hp & 31;
  int base = qr * D_N + h * DH_N + pr * 2;
  if (qr >= S_N) { q16[base] = (_Float16)0.f; q16[base + 1] = (_Float16)0.f; return; }
  float cy = (float)(qr / GRID_N), cx = (float)(qr % GRID_N);
  float ang = (float)T_N * __powf(10000.f, -(float)pr * (1.f / 32.f));
  if (pr < 16) ang += cy * __powf(10000.f, -(float)pr * (1.f / 16.f));
  else         ang += cx * __powf(10000.f, -(float)(pr - 16) * (1.f / 16.f));
  float s, c; __sincosf(ang, &s, &c);
  float q0 = qf[base], q1 = qf[base + 1];
  q16[base]     = (_Float16)((q0 * c - q1 * s) * 0.125f);   // fold 1/sqrt(64)
  q16[base + 1] = (_Float16)((q0 * s + q1 * c) * 0.125f);
}

// ---- flash attention: one wave per (head, 16-query tile) -------------------
__global__ void __launch_bounds__(128)
flash_attn(const _Float16* __restrict__ q16,   // QROWS x 512
           const _Float16* __restrict__ k16,   // NTP x 128  (rope'd)
           const _Float16* __restrict__ vT,    // 128 x NTP  (per-head transposed V)
           _Float16* __restrict__ attn16) {    // QROWS x 512
  __shared__ __align__(16) _Float16 plds[4][16][32];
  int wid  = threadIdx.x >> 5;
  int wave = blockIdx.x * 4 + wid;
  const int QT = QROWS / 16;                  // 57
  if (wave >= HQ_N * QT) return;
  int h = wave / QT, qt = wave % QT;
  int kvh = h >> 2;                           // GQA group of 4
  int lane = threadIdx.x & 31;
  int col = lane & 15;
  int rowbase = (lane >> 4) * 8;
  int qrow0 = qt * 16 + rowbase;

  const _Float16* Qb = q16 + (long)(qt * 16) * D_N + h * DH_N;
  v16h qa0 = load_tile_AB(Qb,      D_N);      // d = 0..31
  v16h qa1 = load_tile_AB(Qb + 32, D_N);      // d = 32..63

  v8f acc0 = {}, acc1 = {}, acc2 = {}, acc3 = {};
  float m[8], l[8];
#pragma unroll
  for (int j = 0; j < 8; ++j) { m[j] = -3.0e38f; l[j] = 0.f; }

  _Float16* myLds = &plds[wid][0][0];
  const _Float16* Vb = vT + (long)(kvh * DH_N) * NTP;

  for (int t0 = 0; t0 < NTP; t0 += 32) {
    const _Float16* Kb = k16 + (long)t0 * 128 + kvh * DH_N;
    v16h kb00 = load_tile_AB(Kb,            128);
    v16h kb01 = load_tile_AB(Kb + 32,       128);
    v16h kb10 = load_tile_AB(Kb + 16 * 128, 128);
    v16h kb11 = load_tile_AB(Kb + 16 * 128 + 32, 128);
    v8f s0 = {}, s1 = {};
    s0 = wmma_f16(qa0, kb00, s0);
    s0 = wmma_f16(qa1, kb01, s0);
    s1 = wmma_f16(qa0, kb10, s1);
    s1 = wmma_f16(qa1, kb11, s1);

    // visibility mask: lane covers tokens t0+col and t0+16+col, rows qrow0+j
    int tokA = t0 + col, tokB = t0 + 16 + col;
    bool liveA = tokA < NTOK, liveB = tokB < NTOK;
    bool histA = (tokA >= 1) & (tokA <= HIST_END);
    bool histB = (tokB >= 1) & (tokB <= HIST_END);
    int cellA = histA ? (tokA - 1) % S_N : -1;
    int cellB = histB ? (tokB - 1) % S_N : -1;

    float p0[8], p1[8];
#pragma unroll
    for (int j = 0; j < 8; ++j) {
      int qr = qrow0 + j;
      bool visA = liveA && (!histA || cellA == qr);
      bool visB = liveB && (!histB || cellB == qr);
      float a = visA ? s0[j] : -3.0e38f;
      float b = visB ? s1[j] : -3.0e38f;
      float mx = redmax16(fmaxf(a, b));       // row max across 16 cols
      float mn = fmaxf(m[j], mx);
      float sc = __expf(m[j] - mn);
      float ea = __expf(a - mn);
      float eb = __expf(b - mn);
      float rs = redsum16(ea + eb);           // row sum of probs
      l[j] = l[j] * sc + rs;
      m[j] = mn;
      acc0[j] *= sc; acc1[j] *= sc; acc2[j] *= sc; acc3[j] *= sc;
      p0[j] = ea; p1[j] = eb;
    }

    // C-layout -> A-layout re-tiling of P through per-wave LDS
#pragma unroll
    for (int j = 0; j < 8; ++j) {
      myLds[(rowbase + j) * 32 + col]      = (_Float16)p0[j];
      myLds[(rowbase + j) * 32 + 16 + col] = (_Float16)p1[j];
    }
    asm volatile("s_wait_dscnt 0" ::: "memory");   // CDNA5 split DS counter wait
    v16h pa  = load_tile_AB(myLds, 32);            // P as 16x32 A operand

    v16h vb0 = load_tile_AB(Vb + t0,             NTP);
    v16h vb1 = load_tile_AB(Vb + 16L * NTP + t0, NTP);
    v16h vb2 = load_tile_AB(Vb + 32L * NTP + t0, NTP);
    v16h vb3 = load_tile_AB(Vb + 48L * NTP + t0, NTP);
    acc0 = wmma_f16(pa, vb0, acc0);
    acc1 = wmma_f16(pa, vb1, acc1);
    acc2 = wmma_f16(pa, vb2, acc2);
    acc3 = wmma_f16(pa, vb3, acc3);
  }

  _Float16* Ob = attn16 + (long)(qt * 16) * D_N + h * DH_N;
#pragma unroll
  for (int j = 0; j < 8; ++j) {
    float inv = 1.f / l[j];
    int r = rowbase + j;
    Ob[(long)r * D_N + col]      = (_Float16)(acc0[j] * inv);
    Ob[(long)r * D_N + 16 + col] = (_Float16)(acc1[j] * inv);
    Ob[(long)r * D_N + 32 + col] = (_Float16)(acc2[j] * inv);
    Ob[(long)r * D_N + 48 + col] = (_Float16)(acc3[j] * inv);
  }
}

// ---- host-side orchestration ----------------------------------------------
extern "C" void kernel_launch(void* const* d_in, const int* in_sizes, int n_in,
                              void* d_out, int out_size, void* d_ws, size_t ws_size,
                              hipStream_t stream) {
  (void)in_sizes; (void)n_in; (void)out_size; (void)ws_size;
  const float* sg   = (const float*)d_in[0];
  const float* hist = (const float*)d_in[1];
  const float* init = (const float*)d_in[2];
  const float* Wq   = (const float*)d_in[3];
  const float* Wk   = (const float*)d_in[4];
  const float* Wv   = (const float*)d_in[5];
  const float* Wo   = (const float*)d_in[6];
  float* out = (float*)d_out;

  char* ws = (char*)d_ws;
  size_t off = 0;
  auto alloc = [&](size_t bytes) -> void* {
    void* p = ws + off;
    off = (off + bytes + 255) & ~(size_t)255;
    return p;
  };
  _Float16* tok16 = (_Float16*)alloc((size_t)NTP * D_N * 2);
  _Float16* wq16  = (_Float16*)alloc((size_t)D_N * D_N * 2);
  _Float16* wk16  = (_Float16*)alloc((size_t)128 * D_N * 2);
  _Float16* wv16  = (_Float16*)alloc((size_t)128 * D_N * 2);
  _Float16* wo16  = (_Float16*)alloc((size_t)D_N * D_N * 2);
  float*    kf32  = (float*)  alloc((size_t)NTP * 128 * 4);
  float*    vf32  = (float*)  alloc((size_t)NTP * 128 * 4);
  float*    qf32  = (float*)  alloc((size_t)QROWS * D_N * 4);
  _Float16* k16   = (_Float16*)alloc((size_t)NTP * 128 * 2);
  _Float16* vT    = (_Float16*)alloc((size_t)128 * NTP * 2);
  _Float16* q16   = (_Float16*)alloc((size_t)QROWS * D_N * 2);
  _Float16* a16   = (_Float16*)alloc((size_t)QROWS * D_N * 2);

  // 1) token packing + weight f16 conversion
  { long n = (long)NTP * D_N;
    pack_tokens<<<(int)((n + 255) / 256), 256, 0, stream>>>(sg, hist, init, tok16); }
  cvt_f32_f16<<<(D_N * D_N + 255) / 256, 256, 0, stream>>>(Wq, wq16, D_N * D_N);
  cvt_f32_f16<<<(128 * D_N + 255) / 256, 256, 0, stream>>>(Wk, wk16, 128 * D_N);
  cvt_f32_f16<<<(128 * D_N + 255) / 256, 256, 0, stream>>>(Wv, wv16, 128 * D_N);
  cvt_f32_f16<<<(D_N * D_N + 255) / 256, 256, 0, stream>>>(Wo, wo16, D_N * D_N);

  // 2) K/V projections over all tokens
  { int tiles = (NTP / 16) * (128 / 16);
    gemm_f16<<<(tiles + 3) / 4, 128, 0, stream>>>(tok16, D_N, wk16, D_N, kf32, 128,
                                                  NTP / 16, 128 / 16, D_N, NTP);
    gemm_f16<<<(tiles + 3) / 4, 128, 0, stream>>>(tok16, D_N, wv16, D_N, vf32, 128,
                                                  NTP / 16, 128 / 16, D_N, NTP); }

  // 3) Q projection (A = current-frame token rows; pad rows are zero)
  { int tiles = (QROWS / 16) * (D_N / 16);
    gemm_f16<<<(tiles + 3) / 4, 128, 0, stream>>>(tok16 + (size_t)CUR_BASE * D_N, D_N,
                                                  wq16, D_N, qf32, D_N,
                                                  QROWS / 16, D_N / 16, D_N, QROWS); }

  // 4) RoPE + f16 pack
  { int n = NTP * 64;
    rope_pack_kv<<<(n + 255) / 256, 256, 0, stream>>>(kf32, vf32, k16, vT); }
  { int n = QROWS * 256;
    rope_pack_q<<<(n + 255) / 256, 256, 0, stream>>>(qf32, q16); }

  // 5) flash attention
  { int waves = HQ_N * (QROWS / 16);
    flash_attn<<<(waves + 3) / 4, 128, 0, stream>>>(q16, k16, vT, a16); }

  // 6) output projection -> d_out (f32, rows 0..899 only)
  { int tiles = (QROWS / 16) * (D_N / 16);
    gemm_f16<<<(tiles + 3) / 4, 128, 0, stream>>>(a16, D_N, wo16, D_N, out, D_N,
                                                  QROWS / 16, D_N / 16, D_N, S_N); }
}
